// TransformerBlockONNX_6116033429976
// MI455X (gfx1250) — matrix operand order repaired and checked by reading, not compiled, not run on hardware
//
#include <hip/hip_runtime.h>
#include <cstdint>

// ---------------- problem constants ----------------
#define B_   2
#define S_   2048
#define D_   1024
#define H_   16
#define HD_  64
#define FF_  4096
#define COND_ 1024
#define WIN_ 256
#define SCALE_ 0.125f
#define EPS_ 1e-5f
#define NINF (-__builtin_inff())

typedef __attribute__((ext_vector_type(16))) __bf16 v16bf;
typedef __attribute__((ext_vector_type(8)))  float  v8f;

// ---- CDNA5 async global->LDS copy (ASYNCcnt), guarded by builtin probe ----
#if __has_builtin(__builtin_amdgcn_global_load_async_to_lds_b128)
#define USE_ASYNC_LDS 1
#else
#define USE_ASYNC_LDS 0
#endif

#if USE_ASYNC_LDS
typedef int v4i __attribute__((vector_size(16)));
typedef __attribute__((address_space(1))) v4i gas_v4i;
typedef __attribute__((address_space(3))) v4i las_v4i;
static __device__ inline void acp16(__bf16* l, const __bf16* g) {
  __builtin_amdgcn_global_load_async_to_lds_b128(
      (gas_v4i*)g, (las_v4i*)l, 0, 0);
}
#if __has_builtin(__builtin_amdgcn_s_wait_asynccnt)
#define ASYNC_DRAIN() __builtin_amdgcn_s_wait_asynccnt(0)
#else
#define ASYNC_DRAIN() asm volatile("s_wait_asynccnt 0x0" ::: "memory")
#endif
#else
#define ASYNC_DRAIN()
#endif

static __device__ inline v8f zero8() {
  v8f z;
#pragma unroll
  for (int i = 0; i < 8; i++) z[i] = 0.f;
  return z;
}

static __device__ inline v8f wmma_bf16(v16bf a, v16bf b, v8f c) {
  return __builtin_amdgcn_wmma_f32_16x16x32_bf16(
      false, a, false, b, (short)0, c, false, false);
}

// load a v16bf fragment as two 16B quads (ds_load_b128 / global_load_b128)
static __device__ inline v16bf ld2q(const __bf16* p0, const __bf16* p1) {
  union { v16bf v; int4 q[2]; } u;
  u.q[0] = *(const int4*)p0;
  u.q[1] = *(const int4*)p1;
  return u.v;
}

// pack 8 f32 -> 8 bf16 and store as one b128
static __device__ inline void st8bf(__bf16* p, const float* f) {
  union { __bf16 h[8]; int4 q; } u;
#pragma unroll
  for (int i = 0; i < 8; i++) u.h[i] = (__bf16)f[i];
  *(int4*)p = u.q;
}

// ---------------- f32 (K,N) -> bf16 transposed (N,K), 32x32 LDS tiles -------
__global__ __launch_bounds__(256) void k_cvt_tr(const float* __restrict__ src,
                                                __bf16* __restrict__ dst,
                                                int K, int N) {
  __shared__ float Ts[32 * 33];
  const int k0 = blockIdx.y * 32, n0 = blockIdx.x * 32;
  const int tid = threadIdx.x;
  const int r = tid >> 3, c4 = (tid & 7) * 4;
  float4 f = *(const float4*)&src[(size_t)(k0 + r) * N + n0 + c4];
  Ts[(c4 + 0) * 33 + r] = f.x;
  Ts[(c4 + 1) * 33 + r] = f.y;
  Ts[(c4 + 2) * 33 + r] = f.z;
  Ts[(c4 + 3) * 33 + r] = f.w;
  __syncthreads();
  union { __bf16 h[4]; uint2 q; } u;
#pragma unroll
  for (int i = 0; i < 4; i++) u.h[i] = (__bf16)Ts[r * 33 + c4 + i];
  *(uint2*)&dst[(size_t)(n0 + r) * K + k0 + c4] = u.q;
}

// ---------------- silu(condition) ----------------
__global__ __launch_bounds__(256) void k_silu(const float* __restrict__ c,
                                              float* __restrict__ sc, int n) {
  int i = blockIdx.x * 256 + threadIdx.x;
  if (i < n) { float v = c[i]; sc[i] = __fdividef(v, 1.f + __expf(-v)); }
}

// ---------------- adaLN modulation: mod = silu(cond) @ w + b ----------------
__global__ __launch_bounds__(256) void k_adaln(const float* __restrict__ sc,
                                               const float* __restrict__ w,
                                               const float* __restrict__ bias,
                                               float* __restrict__ mod) {
  int idx = blockIdx.x * 256 + threadIdx.x;       // B * 3D
  if (idx >= B_ * 3 * D_) return;
  int b = idx / (3 * D_), j = idx % (3 * D_);
  const float* s = sc + b * COND_;
  float acc = 0.f;
  for (int c = 0; c < COND_; c++) acc += s[c] * w[(size_t)c * (3 * D_) + j];
  mod[idx] = acc + bias[j];
}

// ---------------- LayerNorm + (1+scale)*x + shift -> bf16 ----------------
__global__ __launch_bounds__(256) void k_ln_mod(const float* __restrict__ x,
                                                const float* __restrict__ mod,
                                                __bf16* __restrict__ out) {
  int row = blockIdx.x;            // b*S + s
  int b = row >> 11;               // row / S
  int tid = threadIdx.x;
  float4 xv = ((const float4*)(x + (size_t)row * D_))[tid];
  __shared__ float r1[256], r2[256];
  r1[tid] = xv.x + xv.y + xv.z + xv.w;
  r2[tid] = xv.x * xv.x + xv.y * xv.y + xv.z * xv.z + xv.w * xv.w;
  __syncthreads();
  for (int o = 128; o > 0; o >>= 1) {
    if (tid < o) { r1[tid] += r1[tid + o]; r2[tid] += r2[tid + o]; }
    __syncthreads();
  }
  float mean = r1[0] * (1.f / D_);
  float var = r2[0] * (1.f / D_) - mean * mean;
  float rs = rsqrtf(var + EPS_);
  int d = tid * 4;
  const float* mo = mod + (size_t)b * 3 * D_;
  float4 sc4 = *(const float4*)(mo + D_ + d);
  float4 sh4 = *(const float4*)(mo + d);
  union { __bf16 h[4]; uint2 q; } u;
  u.h[0] = (__bf16)((xv.x - mean) * rs * (1.f + sc4.x) + sh4.x);
  u.h[1] = (__bf16)((xv.y - mean) * rs * (1.f + sc4.y) + sh4.y);
  u.h[2] = (__bf16)((xv.z - mean) * rs * (1.f + sc4.z) + sh4.z);
  u.h[3] = (__bf16)((xv.w - mean) * rs * (1.f + sc4.w) + sh4.w);
  *(uint2*)(out + (size_t)row * D_ + d) = u.q;
}

// ---------------- GEMM tile staging (weights pre-transposed (N,K)) ----------
static __device__ inline void stage_A(const __bf16* __restrict__ A, __bf16* Asb,
                                      int tid, int m0, int k0, int K) {
  int row = tid >> 2, ce = (tid & 3) * 8;     // 64 rows x 32 K
  const __bf16* gp = &A[(size_t)(m0 + row) * K + k0 + ce];
  __bf16* lp = &Asb[row * 32 + ce];
#if USE_ASYNC_LDS
  acp16(lp, gp);
#else
  *(int4*)lp = *(const int4*)gp;
#endif
}
static __device__ inline void stage_W(const __bf16* __restrict__ Wt, __bf16* Wsb,
                                      int tid, int bn0, int k0, int K) {
  int col = tid >> 1, koff = (tid & 1) * 16;  // 128 cols x 32 K
  const __bf16* sp = &Wt[(size_t)(bn0 + col) * K + k0 + koff];
  __bf16* lp = &Wsb[col * 32 + koff];
#if USE_ASYNC_LDS
  acp16(lp, sp);
  acp16(lp + 8, sp + 8);
#else
  int4 lo = *(const int4*)sp;
  int4 hi = *(const int4*)(sp + 8);
  *(int4*)lp = lo;
  *(int4*)(lp + 8) = hi;
#endif
}

// ---------------- generic WMMA GEMM: C[MxN] = A[MxK] @ Wt[NxK]^T ------------
// block = 256 threads (8 waves), macro tile 64x128, double-buffered LDS.
// MODE 0: out bf16 = acc (row-major)
// MODE 1: out f32  = res + gate * acc   (gate = mod[b*3N + 2N + n], b = m/S)
// MODE 2: out bf16 transposed (b,h,hd,s) for attention V
template <int MODE>
__global__ __launch_bounds__(256) void k_gemm(const __bf16* __restrict__ A,
                                              const __bf16* __restrict__ Wt,
                                              void* __restrict__ outp,
                                              const float* __restrict__ res,
                                              const float* __restrict__ mod,
                                              int M, int N, int K) {
  __shared__ __align__(16) __bf16 As[2][64 * 32];
  __shared__ __align__(16) __bf16 Ws[2][128 * 32];
  __shared__ __align__(16) float  Cs[16 * 128];
  const int tid = threadIdx.x, lane = tid & 31, wid = tid >> 5;
  const int half = lane >> 4, ln = lane & 15;
  const int m0 = blockIdx.y * 64;
  const int bn0 = blockIdx.x * 128;
  v8f acc[4] = {zero8(), zero8(), zero8(), zero8()};

  stage_A(A, As[0], tid, m0, 0, K);
  stage_W(Wt, Ws[0], tid, bn0, 0, K);
  ASYNC_DRAIN();
  __syncthreads();

  const int cc = wid * 16 + ln;
  const int kb = 16 * half;
  const int ko = 8 * half;
  for (int k0 = 0; k0 < K; k0 += 32) {
    const int cb = (k0 >> 5) & 1;
    if (k0 + 32 < K) {        // issue next tile's copies before compute
      stage_A(A, As[cb ^ 1], tid, m0, k0 + 32, K);
      stage_W(Wt, Ws[cb ^ 1], tid, bn0, k0 + 32, K);
    }
    const __bf16* Asb = As[cb];
    const __bf16* Wsb = Ws[cb];
    v16bf bf = ld2q(&Wsb[cc * 32 + kb], &Wsb[cc * 32 + kb + 8]);
#pragma unroll
    for (int rt = 0; rt < 4; rt++) {
      int m = rt * 16 + ln;
      v16bf af = ld2q(&Asb[m * 32 + kb], &Asb[m * 32 + kb + 8]);
      acc[rt] = wmma_bf16(af, bf, acc[rt]);
    }
    ASYNC_DRAIN();
    __syncthreads();
  }
  // ---------------- epilogue ----------------
  if (MODE == 2) {
    // transposed store: per lane 8 consecutive rows (s) at fixed col -> b128
    int n = bn0 + wid * 16 + ln;
    int hh = n >> 6, hd = n & 63;
    int b = m0 / S_;
    size_t colbase = ((size_t)(b * H_ + hh) * HD_ + hd) * S_ + (m0 - (size_t)b * S_);
#pragma unroll
    for (int rt = 0; rt < 4; rt++) {
      float f[8];
#pragma unroll
      for (int v = 0; v < 8; v++) f[v] = acc[rt][v];
      st8bf((__bf16*)outp + colbase + rt * 16 + 8 * half, f);
    }
    return;
  }
  const int b = m0 / S_;
#pragma unroll
  for (int rt = 0; rt < 4; rt++) {
    __syncthreads();
    {
      int colc = wid * 16 + ln;
#pragma unroll
      for (int v = 0; v < 8; v++) Cs[(v + 8 * half) * 128 + colc] = acc[rt][v];
    }
    __syncthreads();
    // cooperative packed store: 2048 f32, 8 per thread
    int e = tid * 8;
    int r = e >> 7, c = e & 127;
    int m = m0 + rt * 16 + r;
    int n = bn0 + c;
    const float* cp = &Cs[r * 128 + c];
    if (MODE == 0) {
      float f[8];
#pragma unroll
      for (int i = 0; i < 8; i++) f[i] = cp[i];
      st8bf((__bf16*)outp + (size_t)m * N + n, f);
    } else {
      float4 g0 = *(const float4*)&mod[(size_t)b * 3 * N + 2 * N + n];
      float4 g1 = *(const float4*)&mod[(size_t)b * 3 * N + 2 * N + n + 4];
      float4 r0 = *(const float4*)&res[(size_t)m * N + n];
      float4 r1 = *(const float4*)&res[(size_t)m * N + n + 4];
      float4 o0, o1;
      o0.x = r0.x + g0.x * cp[0]; o0.y = r0.y + g0.y * cp[1];
      o0.z = r0.z + g0.z * cp[2]; o0.w = r0.w + g0.w * cp[3];
      o1.x = r1.x + g1.x * cp[4]; o1.y = r1.y + g1.y * cp[5];
      o1.z = r1.z + g1.z * cp[6]; o1.w = r1.w + g1.w * cp[7];
      *(float4*)((float*)outp + (size_t)m * N + n) = o0;
      *(float4*)((float*)outp + (size_t)m * N + n + 4) = o1;
    }
  }
}

// ---------------- fused SwiGLU GEMM: out = silu(A@W1) * (A@W3) -> bf16 ------
__global__ __launch_bounds__(256) void k_gemm_swiglu(const __bf16* __restrict__ A,
                                                     const __bf16* __restrict__ W1t,
                                                     const __bf16* __restrict__ W3t,
                                                     __bf16* __restrict__ out,
                                                     int M, int N, int K) {
  __shared__ __align__(16) __bf16 As[2][64 * 32];
  __shared__ __align__(16) __bf16 Ws1[2][128 * 32];
  __shared__ __align__(16) __bf16 Ws3[2][128 * 32];
  __shared__ __align__(16) float  Cs[16 * 128];
  const int tid = threadIdx.x, lane = tid & 31, wid = tid >> 5;
  const int half = lane >> 4, ln = lane & 15;
  const int m0 = blockIdx.y * 64;
  const int bn0 = blockIdx.x * 128;
  v8f a1[4] = {zero8(), zero8(), zero8(), zero8()};
  v8f a3[4] = {zero8(), zero8(), zero8(), zero8()};

  stage_A(A, As[0], tid, m0, 0, K);
  stage_W(W1t, Ws1[0], tid, bn0, 0, K);
  stage_W(W3t, Ws3[0], tid, bn0, 0, K);
  ASYNC_DRAIN();
  __syncthreads();

  const int cc = wid * 16 + ln;
  const int kb = 16 * half;
  const int ko = 8 * half;
  for (int k0 = 0; k0 < K; k0 += 32) {
    const int cb = (k0 >> 5) & 1;
    if (k0 + 32 < K) {
      stage_A(A, As[cb ^ 1], tid, m0, k0 + 32, K);
      stage_W(W1t, Ws1[cb ^ 1], tid, bn0, k0 + 32, K);
      stage_W(W3t, Ws3[cb ^ 1], tid, bn0, k0 + 32, K);
    }
    const __bf16* Asb = As[cb];
    v16bf b1 = ld2q(&Ws1[cb][cc * 32 + kb], &Ws1[cb][cc * 32 + kb + 8]);
    v16bf b3 = ld2q(&Ws3[cb][cc * 32 + kb], &Ws3[cb][cc * 32 + kb + 8]);
#pragma unroll
    for (int rt = 0; rt < 4; rt++) {
      int m = rt * 16 + ln;
      v16bf af = ld2q(&Asb[m * 32 + kb], &Asb[m * 32 + kb + 8]);
      a1[rt] = wmma_bf16(af, b1, a1[rt]);
      a3[rt] = wmma_bf16(af, b3, a3[rt]);
    }
    ASYNC_DRAIN();
    __syncthreads();
  }
#pragma unroll
  for (int rt = 0; rt < 4; rt++) {
    __syncthreads();
    {
      int colc = wid * 16 + ln;
#pragma unroll
      for (int v = 0; v < 8; v++) {
        float a = a1[rt][v];
        Cs[(v + 8 * half) * 128 + colc] = __fdividef(a, 1.f + __expf(-a)) * a3[rt][v];
      }
    }
    __syncthreads();
    int e = tid * 8;
    int r = e >> 7, c = e & 127;
    int m = m0 + rt * 16 + r;
    int n = bn0 + c;
    float f[8];
#pragma unroll
    for (int i = 0; i < 8; i++) f[i] = Cs[r * 128 + c + i];
    st8bf(out + (size_t)m * N + n, f);
  }
}

// ---------------- RoPE in place on bf16 (B,S,H,HD) ----------------
__global__ __launch_bounds__(256) void k_rope(__bf16* __restrict__ buf,
                                              const float* __restrict__ cs,
                                              const float* __restrict__ sn) {
  int idx = blockIdx.x * 256 + threadIdx.x;     // B*S*H*32
  if (idx >= B_ * S_ * H_ * (HD_ / 2)) return;
  int p = idx & 31;
  int bs = idx >> 9;                             // b*S + s
  int s = bs & (S_ - 1);
  size_t base = (size_t)bs * D_ + ((idx >> 5) & (H_ - 1)) * HD_ + 2 * p;
  union { uint32_t u; __bf16 h[2]; } io;
  io.u = *(const uint32_t*)(buf + base);
  float xr = (float)io.h[0], xi = (float)io.h[1];
  float c = cs[s * 32 + p], ss = sn[s * 32 + p];
  io.h[0] = (__bf16)(xr * c - xi * ss);
  io.h[1] = (__bf16)(xr * ss + xi * c);
  *(uint32_t*)(buf + base) = io.u;
}

// ---------------- windowed causal flash attention -------------------------
// 4 waves / block, each wave owns one 16-query tile. V pre-transposed
// (b,h,hd,s). Per-wave LDS slices; wave-local s_wait_dscnt (no block barrier:
// waves have different window trip counts).
__global__ __launch_bounds__(128) void k_attn(const __bf16* __restrict__ Q,
                                              const __bf16* __restrict__ Kb,
                                              const __bf16* __restrict__ Vt,
                                              __bf16* __restrict__ Ob) {
  const int lane = threadIdx.x & 31, wid = threadIdx.x >> 5;
  const int q0 = blockIdx.x * 64 + wid * 16;
  const int bh = blockIdx.y;
  const int b = bh >> 4, h = bh & 15;
  const int half = lane >> 4, ln = lane & 15;
  __shared__ __align__(16) __bf16 Pl[4][16 * 32];
  __shared__ __align__(16) __bf16 Os[4][16 * 64];
  __bf16* Pw = Pl[wid];
  __bf16* Ow = Os[wid];

  const __bf16* qrow = Q + ((size_t)(b * S_ + q0 + ln) * D_ + h * HD_);
  const int ko = 16 * half;
  v16bf qa0 = ld2q(qrow + ko, qrow + ko + 8);
  v16bf qa1 = ld2q(qrow + 32 + ko, qrow + 40 + ko);

  v8f o0 = zero8(), o1 = zero8(), o2 = zero8(), o3 = zero8();
  float mi[8], li[8];
#pragma unroll
  for (int v = 0; v < 8; v++) { mi[v] = NINF; li[v] = 0.f; }

  const int kb2 = 16 * half;
  int jlo = q0 - WIN_; if (jlo < 0) jlo = 0; jlo &= ~31;
  for (int j0 = jlo; j0 <= q0 + 15; j0 += 32) {
    // ---- scores for 32 keys: two 16x16 tiles ----
    v8f s0 = zero8(), s1 = zero8();
    {
      const __bf16* kr0 = Kb + ((size_t)(b * S_ + j0 + ln) * D_ + h * HD_ + kb2);
      const __bf16* kr1 = Kb + ((size_t)(b * S_ + j0 + 16 + ln) * D_ + h * HD_ + kb2);
      v16bf bk;
      bk = ld2q(kr0, kr0 + 8);       s0 = wmma_bf16(qa0, bk, s0);
      bk = ld2q(kr0 + 32, kr0 + 40); s0 = wmma_bf16(qa1, bk, s0);
      bk = ld2q(kr1, kr1 + 8);       s1 = wmma_bf16(qa0, bk, s1);
      bk = ld2q(kr1 + 32, kr1 + 40); s1 = wmma_bf16(qa1, bk, s1);
    }
    // ---- online softmax (rows live in 16-lane halves) ----
    float al[8], p0[8], p1[8];
#pragma unroll
    for (int v = 0; v < 8; v++) {
      int i = q0 + v + 8 * half;
      int jA = j0 + ln, jB = jA + 16;
      float x0 = (jA <= i && (i - jA) <= WIN_) ? s0[v] * SCALE_ : NINF;
      float x1 = (jB <= i && (i - jB) <= WIN_) ? s1[v] * SCALE_ : NINF;
      float rmax = fmaxf(x0, x1);
      for (int off = 8; off >= 1; off >>= 1) rmax = fmaxf(rmax, __shfl_xor(rmax, off, 32));
      float mn = fmaxf(mi[v], rmax);
      float a, e0, e1;
      if (mn == NINF) { a = 1.f; e0 = 0.f; e1 = 0.f; }
      else {
        a = (mi[v] == NINF) ? 0.f : __expf(mi[v] - mn);
        e0 = (x0 == NINF) ? 0.f : __expf(x0 - mn);
        e1 = (x1 == NINF) ? 0.f : __expf(x1 - mn);
      }
      float rs = e0 + e1;
      for (int off = 8; off >= 1; off >>= 1) rs += __shfl_xor(rs, off, 32);
      li[v] = li[v] * a + rs;
      mi[v] = mn;
      al[v] = a; p0[v] = e0; p1[v] = e1;
    }
#pragma unroll
    for (int v = 0; v < 8; v++) { o0[v] *= al[v]; o1[v] *= al[v]; o2[v] *= al[v]; o3[v] *= al[v]; }
    // ---- P through LDS into A-fragment layout (row-major 16x32) ----
#pragma unroll
    for (int v = 0; v < 8; v++) {
      int m = v + 8 * half;
      Pw[m * 32 + ln] = (__bf16)p0[v];
      Pw[m * 32 + 16 + ln] = (__bf16)p1[v];
    }
    asm volatile("s_wait_dscnt 0" ::: "memory");
    v16bf pa = ld2q(&Pw[ln * 32 + ko], &Pw[ln * 32 + ko + 8]);
    // ---- O += P @ V  (V transposed: contiguous keys per lane) ----
#pragma unroll
    for (int nc = 0; nc < 4; nc++) {
      const __bf16* vp = Vt + ((size_t)(b * H_ + h) * HD_ + nc * 16 + ln) * S_ + j0 + kb2;
      v16bf bv = ld2q(vp, vp + 8);
      v8f* op = (nc == 0) ? &o0 : (nc == 1) ? &o1 : (nc == 2) ? &o2 : &o3;
      *op = wmma_bf16(pa, bv, *op);
    }
  }
  // ---- normalize, stage to LDS, packed coalesced store ----
#pragma unroll
  for (int v = 0; v < 8; v++) {
    float inv = (li[v] > 0.f) ? __fdividef(1.f, li[v]) : 0.f;
    int m = v + 8 * half;
    Ow[m * 64 + 0 + ln]  = (__bf16)(o0[v] * inv);
    Ow[m * 64 + 16 + ln] = (__bf16)(o1[v] * inv);
    Ow[m * 64 + 32 + ln] = (__bf16)(o2[v] * inv);
    Ow[m * 64 + 48 + ln] = (__bf16)(o3[v] * inv);
  }
  asm volatile("s_wait_dscnt 0" ::: "memory");
#pragma unroll
  for (int t = 0; t < 4; t++) {
    int c = t * 32 + lane;           // 128 b128-chunks of 8 bf16
    int row = c >> 3, col = (c & 7) * 8;
    int4 val = *(const int4*)&Ow[row * 64 + col];
    *(int4*)(Ob + (size_t)(b * S_ + q0 + row) * D_ + h * HD_ + col) = val;
  }
}

// ---------------- host launcher ----------------
extern "C" void kernel_launch(void* const* d_in, const int* in_sizes, int n_in,
                              void* d_out, int out_size, void* d_ws, size_t ws_size,
                              hipStream_t stream) {
  (void)in_sizes; (void)n_in; (void)out_size; (void)ws_size;
  const float* x    = (const float*)d_in[0];
  const float* rc   = (const float*)d_in[1];
  const float* rs   = (const float*)d_in[2];
  const float* cond = (const float*)d_in[3];
  const float* wq   = (const float*)d_in[4];
  const float* wk   = (const float*)d_in[5];
  const float* wv   = (const float*)d_in[6];
  const float* wo   = (const float*)d_in[7];
  const float* aaw  = (const float*)d_in[8];
  const float* aab  = (const float*)d_in[9];
  const float* afw  = (const float*)d_in[10];
  const float* afb  = (const float*)d_in[11];
  const float* w1   = (const float*)d_in[12];
  const float* w3   = (const float*)d_in[13];
  const float* w2   = (const float*)d_in[14];
  float* out = (float*)d_out;

  char* ws = (char*)d_ws;
  const size_t MB = 1ull << 20;
  __bf16* wqT  = (__bf16*)(ws + 0 * MB);    // (D,D)   [n][k]
  __bf16* wkT  = (__bf16*)(ws + 2 * MB);
  __bf16* wvT  = (__bf16*)(ws + 4 * MB);
  __bf16* woT  = (__bf16*)(ws + 6 * MB);
  __bf16* w1T  = (__bf16*)(ws + 8 * MB);    // (FF,D)
  __bf16* w3T  = (__bf16*)(ws + 16 * MB);   // (FF,D)
  __bf16* w2T  = (__bf16*)(ws + 24 * MB);   // (D,FF)
  float*  sc    = (float*)(ws + 32 * MB);
  float*  mod_a = (float*)(ws + 32 * MB + (64 << 10));
  float*  mod_f = (float*)(ws + 32 * MB + (128 << 10));
  __bf16* xn    = (__bf16*)(ws + 34 * MB);
  __bf16* qb    = (__bf16*)(ws + 42 * MB);
  __bf16* kb    = (__bf16*)(ws + 50 * MB);
  __bf16* vt    = (__bf16*)(ws + 58 * MB);   // V transposed (b,h,hd,s)
  __bf16* attnb = (__bf16*)(ws + 66 * MB);
  float*  hbuf  = (float*)(ws + 74 * MB);
  __bf16* xn2   = (__bf16*)(ws + 90 * MB);
  __bf16* ff1   = (__bf16*)(ws + 42 * MB);   // aliases qb..attnb (dead by then)

  const int M = B_ * S_;

  // 1) weights -> bf16, transposed to (N,K)
  auto cvtT = [&](const float* s, __bf16* d, int K, int N) {
    dim3 g(N / 32, K / 32);
    k_cvt_tr<<<g, 256, 0, stream>>>(s, d, K, N);
  };
  cvtT(wq, wqT, D_, D_);  cvtT(wk, wkT, D_, D_);
  cvtT(wv, wvT, D_, D_);  cvtT(wo, woT, D_, D_);
  cvtT(w1, w1T, D_, FF_); cvtT(w3, w3T, D_, FF_); cvtT(w2, w2T, FF_, D_);

  // 2) adaLN modulation vectors
  k_silu<<<(B_ * COND_ + 255) / 256, 256, 0, stream>>>(cond, sc, B_ * COND_);
  k_adaln<<<(B_ * 3 * D_ + 255) / 256, 256, 0, stream>>>(sc, aaw, aab, mod_a);
  k_adaln<<<(B_ * 3 * D_ + 255) / 256, 256, 0, stream>>>(sc, afw, afb, mod_f);

  // 3) LN + modulate -> bf16
  k_ln_mod<<<M, 256, 0, stream>>>(x, mod_a, xn);

  // 4) QKV projections (WMMA); V written transposed
  dim3 gD(D_ / 128, M / 64);
  k_gemm<0><<<gD, 256, 0, stream>>>(xn, wqT, qb, nullptr, nullptr, M, D_, D_);
  k_gemm<0><<<gD, 256, 0, stream>>>(xn, wkT, kb, nullptr, nullptr, M, D_, D_);
  k_gemm<2><<<gD, 256, 0, stream>>>(xn, wvT, vt, nullptr, nullptr, M, D_, D_);

  // 5) RoPE on Q and K (in place)
  int nr = B_ * S_ * H_ * (HD_ / 2);
  k_rope<<<(nr + 255) / 256, 256, 0, stream>>>(qb, rc, rs);
  k_rope<<<(nr + 255) / 256, 256, 0, stream>>>(kb, rc, rs);

  // 6) windowed causal attention
  dim3 gA(S_ / 64, B_ * H_);
  k_attn<<<gA, 128, 0, stream>>>(qb, kb, vt, attnb);

  // 7) output projection + gated residual: h = x + gate_a * (attn @ wo)
  k_gemm<1><<<gD, 256, 0, stream>>>(attnb, woT, hbuf, x, mod_a, M, D_, D_);

  // 8) LN2 + modulate -> bf16
  k_ln_mod<<<M, 256, 0, stream>>>(hbuf, mod_f, xn2);

  // 9) fused SwiGLU: ff1 = silu(xn2@w1) * (xn2@w3)
  dim3 gF(FF_ / 128, M / 64);
  k_gemm_swiglu<<<gF, 256, 0, stream>>>(xn2, w1T, w3T, ff1, M, FF_, D_);

  // 10) down-proj + gated residual: out = h + gate_f * (ff1 @ w2)
  k_gemm<1><<<gD, 256, 0, stream>>>(ff1, w2T, out, hbuf, mod_f, M, D_, FF_);
}
